// UnfactedFsBeamformer_67808943669550
// MI455X (gfx1250) — matrix operand order repaired and checked by reading, not compiled, not run on hardware
//
#include <hip/hip_runtime.h>
#include <math.h>

// ---------------------------------------------------------------------------
// UnfactedFsBeamformer on gfx1250 (MI455X, wave32, WMMA).
//
//   f[n,p,k,t] = sum_{c,tap} W[c*256+p, tap] * x[n, c, t*160 + k + tap]
//   out[n,p,t] = log(max(0, max_k f) + 0.01)
//
// One workgroup per (n,t): GEMM M=256 (p) x N=192 (k, padded from 161)
// x K=1664 ((c,tap), taps padded 400->416 with zero weights), computed with
// v_wmma_f32_16x16x32_f16 (f16 inputs, f32 accumulation).
//
// LDS:
//   xpk : x window stored as packed f16 PAIRS xpk[i]=(h(x[i]),h(x[i+1])) so
//         each B-fragment dword (ISA K-pair 2v,2v+1) is one aligned b32 load.
//         Written ONCE before the K loop -> B loads need no barriers, so B
//         fragments are software-pipelined (next frag in flight during WMMA,
//         including across chunk boundaries).
//   wst : double-buffered 256x32 f16 chunk of W; A fragments are two aligned
//         ds_load_b128 per lane in contiguous K order (k = 16h + s, matches B).
// ---------------------------------------------------------------------------

typedef __attribute__((ext_vector_type(16))) _Float16 v16h;
typedef __attribute__((ext_vector_type(8)))  float    v8f;
typedef __attribute__((ext_vector_type(2)))  __fp16   fp16x2;  // cvt_pkrtz result type

#define TAPS   400
#define NCH    4
#define NCHUNK 52          // 4 channels * 13 chunks of 32 (416 padded taps)
#define WIN    560
#define HOP    160
#define T_FR   497
#define P_BM   256
#define KOUT   161         // valid k columns (560-400+1)
#define XWL    608         // padded x-window length per channel
#define NJ     6           // 16-wide N tiles per wave (2 wave-groups -> 192)

union AF { v16h h; uint4 q[2]; };
union BF { v16h h; uint  u[8]; };

__device__ __forceinline__ uint pkh(float a, float b) {
  union { fp16x2 h; uint u; } c;
  c.h = __builtin_amdgcn_cvt_pkrtz(a, b);   // v_cvt_pk_rtz_f16_f32
  return c.u;
}

__global__ __launch_bounds__(512)
void fs_beam_kernel(const float* __restrict__ xg,
                    const float* __restrict__ Wg,
                    float* __restrict__ out) {
  __shared__ uint  xpk[NCH * XWL];       //  9.7 KB packed f16 pairs of x window
  __shared__ uint  wst[2][P_BM * 16];    // 32.0 KB double-buffered W chunk (f16)
  __shared__ float pmax[2][P_BM];        //  2.0 KB cross-wave k-max exchange

  const int t    = blockIdx.x;           // frame
  const int bn   = blockIdx.y;           // batch
  const int tid  = threadIdx.x;
  const int lane = tid & 31;
  const int wave = tid >> 5;             // 16 waves
  const int wm   = wave >> 1;            // p block [32*wm, 32*wm+32)
  const int wn   = wave & 1;             // k-tile group: j = 6*wn + jj
  const int l15  = lane & 15;
  const bool lo  = lane < 16;

  // ---- stage x window as packed f16 pairs (once) --------------------------
  for (int it = 0; it < 5; ++it) {
    int idx = tid + it * 512;
    if (idx < NCH * XWL) {
      int c = idx / XWL, i = idx - c * XWL;
      int sbase = (bn * NCH + c) * 80000 + t * HOP;
      float a = (i     < WIN) ? xg[sbase + i]     : 0.f;
      float b = (i + 1 < WIN) ? xg[sbase + i + 1] : 0.f;
      xpk[idx] = pkh(a, b);
    }
  }

  // ---- W chunk staging: thread -> row p = tid>>1, 16-tap half = (tid&1)*16
  const int sp = tid >> 1;
  const int sh = (tid & 1) << 4;
  auto stageW = [&](int kc, int buf) {
    int c     = kc / 13;
    int tapp0 = (kc - c * 13) * 32;
    uint* dst = &wst[buf][sp * 16 + (sh >> 1)];
    if (tapp0 + sh <= TAPS - 16) {       // chunk halves are all-valid or all-pad
      const float* wr = Wg + (c * P_BM + sp) * TAPS + tapp0 + sh;
      float4 q0 = *(const float4*)(wr + 0);
      float4 q1 = *(const float4*)(wr + 4);
      float4 q2 = *(const float4*)(wr + 8);
      float4 q3 = *(const float4*)(wr + 12);
      uint4 d0, d1;
      d0.x = pkh(q0.x, q0.y); d0.y = pkh(q0.z, q0.w);
      d0.z = pkh(q1.x, q1.y); d0.w = pkh(q1.z, q1.w);
      d1.x = pkh(q2.x, q2.y); d1.y = pkh(q2.z, q2.w);
      d1.z = pkh(q3.x, q3.y); d1.w = pkh(q3.z, q3.w);
      *(uint4*)(dst + 0) = d0;           // ds_store_b128 x2
      *(uint4*)(dst + 4) = d1;
    } else {
      uint4 z = {0u, 0u, 0u, 0u};
      *(uint4*)(dst + 0) = z;
      *(uint4*)(dst + 4) = z;
    }
  };

  // B-fragment gather: lanes 0-15 hold K 0-15 (col = l15), lanes 16-31 hold
  // K 16-31; dword v = samples (s+2v, s+2v+1) = xpk[s+2v].
  const int bOff = (lo ? 0 : 16) + l15 + wn * (NJ * 16);
  auto loadB = [&](int c, int s) {
    const uint* xc = &xpk[c * XWL + s];
    BF b;
#pragma unroll
    for (int v = 0; v < 8; ++v) b.u[v] = xc[2 * v];
    return b;
  };

  stageW(0, 0);
  __syncthreads();

  v8f acc[2][NJ];
#pragma unroll
  for (int mi = 0; mi < 2; ++mi)
#pragma unroll
    for (int jj = 0; jj < NJ; ++jj)
      acc[mi][jj] = (v8f){0.f, 0.f, 0.f, 0.f, 0.f, 0.f, 0.f, 0.f};

  // ---- main K loop ---------------------------------------------------------
  // Per chunk: stage next W chunk (global->LDS), load 2 A frags, then 6x
  // (prefetch next B frag, 2 WMMAs on current). B prefetch crosses the chunk
  // boundary since xpk is immutable -> waits become s_wait_dscnt 8, not 0.
  BF bcur = loadB(0, bOff);              // kc=0, jj=0 (tapp0 = 0)

  for (int kc = 0; kc < NCHUNK; ++kc) {
    const int cur = kc & 1;
    if (kc + 1 < NCHUNK) stageW(kc + 1, cur ^ 1);

    const int c     = kc / 13;
    const int tapp0 = (kc - c * 13) * 32;
    const int kn    = (kc + 1 < NCHUNK) ? kc + 1 : kc;   // dummy on last chunk
    const int cN    = kn / 13;
    const int tappN = (kn - cN * 13) * 32;

    // A fragments: contiguous K layout (matches B). lanes 0-15: K 0-15,
    // lanes 16-31: K 16-31, pairs packed per dword.
    AF a[2];
#pragma unroll
    for (int mi = 0; mi < 2; ++mi) {
      const uint* row = &wst[cur][(wm * 32 + mi * 16 + l15) * 16];
      a[mi].q[0] = *(const uint4*)(row + (lo ? 0 : 8));    // ds_load_b128
      a[mi].q[1] = *(const uint4*)(row + (lo ? 4 : 12));
    }

#pragma unroll
    for (int jj = 0; jj < NJ; ++jj) {
      // prefetch next B fragment (same chunk, or jj=0 of next chunk)
      BF bnext = (jj + 1 < NJ) ? loadB(c, tapp0 + bOff + (jj + 1) * 16)
                               : loadB(cN, tappN + bOff);
      acc[0][jj] = __builtin_amdgcn_wmma_f32_16x16x32_f16(
          false, a[0].h, false, bcur.h, (short)0, acc[0][jj], false, false);
      acc[1][jj] = __builtin_amdgcn_wmma_f32_16x16x32_f16(
          false, a[1].h, false, bcur.h, (short)0, acc[1][jj], false, false);
      bcur = bnext;
    }
    __syncthreads();   // next chunk's wst writes complete for everyone
  }

  // ---- epilogue: relu(max_k) via lane-local max + 16-lane butterfly -------
  // D layout: VGPR r, lanes 0-15 -> p = base + r, lanes 16-31 -> p = base+8+r,
  // col k = 16*j + l15.
#pragma unroll
  for (int mi = 0; mi < 2; ++mi) {
#pragma unroll
    for (int r = 0; r < 8; ++r) {
      float m = 0.f;                      // relu identity
#pragma unroll
      for (int jj = 0; jj < NJ; ++jj) {
        int k = (wn * NJ + jj) * 16 + l15;
        float v = acc[mi][jj][r];
        if (k < KOUT) m = fmaxf(m, v);
      }
#pragma unroll
      for (int msk = 1; msk <= 8; msk <<= 1)
        m = fmaxf(m, __shfl_xor(m, msk, 32));
      if (l15 == 0)
        pmax[wn][wm * 32 + mi * 16 + r + (lo ? 0 : 8)] = m;
    }
  }
  __syncthreads();

  if (tid < P_BM) {
    float v = fmaxf(pmax[0][tid], pmax[1][tid]);
    out[(bn * P_BM + tid) * T_FR + t] = logf(v + 0.01f);
  }
}

extern "C" void kernel_launch(void* const* d_in, const int* in_sizes, int n_in,
                              void* d_out, int out_size, void* d_ws, size_t ws_size,
                              hipStream_t stream) {
  const float* xg = (const float*)d_in[0];   // (4, 4, 80000) f32
  const float* Wg = (const float*)d_in[1];   // (1024, 1, 400, 1) f32
  float* out = (float*)d_out;                // (4, 256, 497) f32
  dim3 grid(T_FR, 4, 1);
  fs_beam_kernel<<<grid, 512, 0, stream>>>(xg, Wg, out);
}
